// GResBlockMeanConv_76879914598739
// MI455X (gfx1250) — compile-verified, matrix operand order checked by reading, not run on hardware
//
#include <hip/hip_runtime.h>
#include <math.h>

typedef __bf16 bf16_t;
typedef __attribute__((ext_vector_type(16))) __bf16 v16bf;
typedef __attribute__((ext_vector_type(8)))  __bf16 v8bf;
typedef __attribute__((ext_vector_type(8)))  float  v8f;

// ---------------------------------------------------------------- utilities

__device__ __forceinline__ void atomicMaxF32(float* addr, float val) {
    // sign-aware int-bits max: init value must be -inf
    if (val >= 0.0f) atomicMax((int*)addr, __float_as_int(val));
    else             atomicMin((unsigned int*)addr, __float_as_uint(val));
}

__global__ void fill_f32(float* __restrict__ p, float v, size_t n) {
    size_t i = (size_t)blockIdx.x * blockDim.x + threadIdx.x;
    size_t s = (size_t)gridDim.x * blockDim.x;
    for (; i < n; i += s) p[i] = v;
}

__global__ void cvt_bf16_kernel(const float* __restrict__ s, bf16_t* __restrict__ d, size_t n) {
    size_t i = (size_t)blockIdx.x * blockDim.x + threadIdx.x;
    size_t st = (size_t)gridDim.x * blockDim.x;
    for (; i < n; i += st) d[i] = (bf16_t)s[i];
}

// ------------------------------------------------------- small prep (once)
// W1t[n,k] = bf16(W1[k,n])  (256x128), W2t[n,k] = bf16(W2[k,n]) (128x256)
// P[0..3]  = layer1 edge-attn coefs P1[h*2+d], P[4..5] = layer2 coefs P2[d]
__global__ void prep_kernel(const float* __restrict__ W1, const float* __restrict__ We1,
                            const float* __restrict__ att_e1,
                            const float* __restrict__ W2, const float* __restrict__ We2,
                            const float* __restrict__ att_e2,
                            bf16_t* __restrict__ W1t, bf16_t* __restrict__ W2t,
                            float* __restrict__ P) {
    int t = threadIdx.x;
    for (int i = t; i < 256 * 128; i += blockDim.x) {
        int n = i >> 7, k = i & 127;
        W1t[i] = (bf16_t)W1[k * 256 + n];
    }
    for (int i = t; i < 128 * 256; i += blockDim.x) {
        int n = i >> 8, k = i & 255;
        W2t[i] = (bf16_t)W2[k * 128 + n];
    }
    if (t < 4) {                 // P1[h][d]: We1 is (2,256), att_e1 is (2,128)
        int h = t >> 1, d = t & 1;
        float s = 0.f;
        for (int c = 0; c < 128; ++c) s += We1[d * 256 + h * 128 + c] * att_e1[h * 128 + c];
        P[t] = s;
    } else if (t < 6) {          // P2[d]: We2 is (2,128), att_e2 is (128)
        int d = t - 4;
        float s = 0.f;
        for (int c = 0; c < 128; ++c) s += We2[d * 128 + c] * att_e2[c];
        P[4 + d] = s;
    }
}

// ------------------------------------------- degree + self-loop mean attrs
__global__ void deg_easum_kernel(const int* __restrict__ dst0, const float* __restrict__ ea,
                                 float* __restrict__ deg, float* __restrict__ easum, int E) {
    int e = blockIdx.x * blockDim.x + threadIdx.x;
    if (e >= E) return;
    int d = dst0[e];
    atomicAdd(&deg[d], 1.0f);
    atomicAdd(&easum[2 * d + 0], ea[2 * e + 0]);
    atomicAdd(&easum[2 * d + 1], ea[2 * e + 1]);
}

__global__ void mean_ea_kernel(const float* __restrict__ deg, const float* __restrict__ easum,
                               float* __restrict__ meanea, int N) {
    int i = blockIdx.x * blockDim.x + threadIdx.x;   // over 2N elements
    if (i >= 2 * N) return;
    float dg = deg[i >> 1];
    meanea[i] = easum[i] / (dg < 1.0f ? 1.0f : dg);
}

// --------------------------------------------------------- WMMA bf16 GEMM
// C[M,N] = A[M,K] * B[K,N],  A bf16 row-major, Bt = B^T bf16 [N,K], C f32.
// One wave per 16x16 C tile; K stepped by 32 via v_wmma_f32_16x16x32_bf16.
__global__ void gemm_bf16_wmma(const bf16_t* __restrict__ A, const bf16_t* __restrict__ Bt,
                               float* __restrict__ C, int M, int N, int K) {
    const int lane = threadIdx.x & 31;
    const int wid  = blockIdx.x * (blockDim.x >> 5) + (threadIdx.x >> 5);
    const int ntn  = N >> 4;
    const int tiles = (M >> 4) * ntn;
    if (wid >= tiles) return;
    const int tm = wid / ntn, tn = wid % ntn;
    const int half = lane >> 4, l16 = lane & 15;

    const bf16_t* Ar = A  + (size_t)(tm * 16 + l16) * K;
    const bf16_t* Br = Bt + (size_t)(tn * 16 + l16) * K;

    v8f acc = {};
    for (int k0 = 0; k0 < K; k0 += 32) {
        // 16-bit A 16x32 layout: lane half selects K octet pairs {h*8.., 16+h*8..}
        v8bf alo = *(const v8bf*)(Ar + k0 + half * 8);
        v8bf ahi = *(const v8bf*)(Ar + k0 + 16 + half * 8);
        v8bf blo = *(const v8bf*)(Br + k0 + half * 8);
        v8bf bhi = *(const v8bf*)(Br + k0 + 16 + half * 8);
        v16bf a, b;
#pragma unroll
        for (int i = 0; i < 8; ++i) {
            a[i] = alo[i]; a[8 + i] = ahi[i];
            b[i] = blo[i]; b[8 + i] = bhi[i];
        }
        acc = __builtin_amdgcn_wmma_f32_16x16x32_bf16(false, a, false, b,
                                                      (short)0, acc, false, false);
    }
    // C/D layout: VGPR v -> row v + 8*half, col = lane%16
    float* Cr = C + (size_t)(tm * 16) * N + tn * 16 + l16;
#pragma unroll
    for (int v = 0; v < 8; ++v) Cr[(size_t)(v + half * 8) * N] = acc[v];
}

// ---------------------------------------------- per-node attention scalars
__global__ void att_node_kernel(const float* __restrict__ xh, const float* __restrict__ att_s,
                                const float* __restrict__ att_d, float* __restrict__ asrc,
                                float* __restrict__ adst, int N, int H, int C) {
    int t = blockIdx.x * blockDim.x + threadIdx.x;
    if (t >= N * H) return;
    int n = t / H, h = t % H;
    const float* xr = xh + (size_t)n * H * C + h * C;
    float ss = 0.f, sd = 0.f;
    for (int c = 0; c < C; ++c) {
        float v = xr[c];
        ss += v * att_s[h * C + c];
        sd += v * att_d[h * C + c];
    }
    asrc[t] = ss;
    adst[t] = sd;
}

// ------------------------------------------------------- edge-wise kernels
__device__ __forceinline__ void edge_ends(int e, int E, const int* src0, const int* dst0,
                                          int& s, int& d) {
    if (e < E) { s = src0[e]; d = dst0[e]; }
    else       { s = d = e - E; }
}

__global__ void edge_alpha_max_kernel(const int* __restrict__ src0, const int* __restrict__ dst0,
                                      const float* __restrict__ ea, const float* __restrict__ meanea,
                                      const float* __restrict__ asrc, const float* __restrict__ adst,
                                      const float* __restrict__ P, float* __restrict__ alpha,
                                      float* __restrict__ amax, int E, int E2, int H) {
    int t = blockIdx.x * blockDim.x + threadIdx.x;
    if (t >= E2 * H) return;
    int e = t / H, h = t % H;
    int s, d; edge_ends(e, E, src0, dst0, s, d);
    float e0, e1;
    if (e < E) { e0 = ea[2 * e]; e1 = ea[2 * e + 1]; }
    else       { int n = e - E; e0 = meanea[2 * n]; e1 = meanea[2 * n + 1]; }
    float a = asrc[s * H + h] + adst[d * H + h] + e0 * P[2 * h] + e1 * P[2 * h + 1];
    a = a > 0.f ? a : 0.2f * a;      // leaky relu
    alpha[t] = a;
    atomicMaxF32(&amax[d * H + h], a);
}

__global__ void edge_exp_sum_kernel(const int* __restrict__ dst0, float* __restrict__ alpha,
                                    const float* __restrict__ amax, float* __restrict__ denom,
                                    int E, int E2, int H) {
    int t = blockIdx.x * blockDim.x + threadIdx.x;
    if (t >= E2 * H) return;
    int e = t / H, h = t % H;
    int d = (e < E) ? dst0[e] : (e - E);
    float ex = __expf(alpha[t] - amax[d * H + h]);
    alpha[t] = ex;                   // in-place: alpha buffer now holds exp
    atomicAdd(&denom[d * H + h], ex);
}

// one block per edge, one thread per output channel -> coalesced gather
__global__ void edge_aggregate_kernel(const int* __restrict__ src0, const int* __restrict__ dst0,
                                      const float* __restrict__ xh, const float* __restrict__ ex,
                                      const float* __restrict__ denom, float* __restrict__ outacc,
                                      int E, int H, int C) {
    int e = blockIdx.x;
    int j = threadIdx.x;             // 0 .. H*C-1
    int s, d; edge_ends(e, E, src0, dst0, s, d);
    int h = j / C;
    float w = ex[e * H + h] / (denom[d * H + h] + 1e-16f);
    atomicAdd(&outacc[(size_t)d * H * C + j], xh[(size_t)s * H * C + j] * w);
}

// --------------------------------------------------------------- epilogues
__global__ void bias_relu_bf16_kernel(const float* __restrict__ acc, const float* __restrict__ b,
                                      bf16_t* __restrict__ hb, size_t n, int C) {
    size_t i = (size_t)blockIdx.x * blockDim.x + threadIdx.x;
    size_t st = (size_t)gridDim.x * blockDim.x;
    for (; i < n; i += st) {
        float v = acc[i] + b[i % C];
        hb[i] = (bf16_t)(v > 0.f ? v : 0.f);
    }
}

__global__ void bias_add_kernel(float* __restrict__ acc, const float* __restrict__ b,
                                size_t n, int C) {
    size_t i = (size_t)blockIdx.x * blockDim.x + threadIdx.x;
    size_t st = (size_t)gridDim.x * blockDim.x;
    for (; i < n; i += st) acc[i] = acc[i] + b[i % C];
}

__global__ void scatter_sum_kernel(const int* __restrict__ src0, const int* __restrict__ dst0,
                                   const float* __restrict__ h, float* __restrict__ acc, int C) {
    int e = blockIdx.x;
    int c = threadIdx.x;
    int s = src0[e], d = dst0[e];
    atomicAdd(&acc[(size_t)d * C + c], h[(size_t)s * C + c]);
}

__global__ void final_kernel(const float* __restrict__ acc, const float* __restrict__ deg,
                             const float* __restrict__ x, float* __restrict__ out,
                             size_t n, int C) {
    size_t i = (size_t)blockIdx.x * blockDim.x + threadIdx.x;
    size_t st = (size_t)gridDim.x * blockDim.x;
    for (; i < n; i += st) {
        float dg = deg[i / C];
        dg = dg < 1.f ? 1.f : dg;
        float v = acc[i] / dg + x[i];
        out[i] = v > 0.f ? v : 0.f;
    }
}

// ================================================================= launch

extern "C" void kernel_launch(void* const* d_in, const int* in_sizes, int n_in,
                              void* d_out, int out_size, void* d_ws, size_t ws_size,
                              hipStream_t stream) {
    (void)n_in; (void)out_size; (void)ws_size;
    const float* x      = (const float*)d_in[0];
    const int*   ei     = (const int*)  d_in[1];
    const float* ea     = (const float*)d_in[2];
    const float* W1     = (const float*)d_in[3];
    const float* att_s1 = (const float*)d_in[4];
    const float* att_d1 = (const float*)d_in[5];
    const float* We1    = (const float*)d_in[6];
    const float* att_e1 = (const float*)d_in[7];
    const float* b1     = (const float*)d_in[8];
    const float* W2     = (const float*)d_in[9];
    const float* att_s2 = (const float*)d_in[10];
    const float* att_d2 = (const float*)d_in[11];
    const float* We2    = (const float*)d_in[12];
    const float* att_e2 = (const float*)d_in[13];
    const float* b2     = (const float*)d_in[14];
    float* out = (float*)d_out;

    const int N  = in_sizes[0] / 128;   // 20000
    const int E  = in_sizes[1] / 2;     // 400000
    const int E2 = E + N;               // with self loops
    const int* src0 = ei;
    const int* dst0 = ei + E;

    // ---- workspace carve (256B aligned), with buffer reuse
    char* p = (char*)d_ws;
    auto alloc = [&](size_t bytes) -> void* {
        void* r = (void*)p;
        p += (bytes + 255) & ~(size_t)255;
        return r;
    };
    float*  deg    = (float*) alloc((size_t)N * 4);
    float*  easum  = (float*) alloc((size_t)N * 2 * 4);
    float*  meanea = (float*) alloc((size_t)N * 2 * 4);
    bf16_t* xb     = (bf16_t*)alloc((size_t)N * 128 * 2);
    bf16_t* W1t    = (bf16_t*)alloc((size_t)256 * 128 * 2);
    bf16_t* W2t    = (bf16_t*)alloc((size_t)128 * 256 * 2);
    float*  P      = (float*) alloc(8 * 4);
    float*  xh1    = (float*) alloc((size_t)N * 256 * 4);
    float*  asrc1  = (float*) alloc((size_t)N * 2 * 4);
    float*  adst1  = (float*) alloc((size_t)N * 2 * 4);
    float*  alpha1 = (float*) alloc((size_t)E2 * 2 * 4);
    float*  amax1  = (float*) alloc((size_t)N * 2 * 4);
    float*  denom1 = (float*) alloc((size_t)N * 2 * 4);
    float*  out1   = (float*) alloc((size_t)N * 256 * 4);
    bf16_t* hb     = (bf16_t*)alloc((size_t)N * 256 * 2);
    // reuse (sequentially safe):
    float*  xh2    = out1;            // [N,128] after out1 consumed into hb
    float*  out2   = xh1;             // [N,128] layer-2 accumulator / h2
    float*  acc    = (float*)hb;      // [N,128] f32 == N*256 bf16 bytes
    float*  alpha2 = alpha1;
    float*  amax2  = amax1;
    float*  denom2 = denom1;
    float*  asrc2  = asrc1;
    float*  adst2  = adst1;

    const int B = 256;
    auto g1 = [](size_t n, int b) { return (int)((n + (size_t)b - 1) / b); };

    // ---- degree + self-loop mean edge attr
    fill_f32<<<g1(N, B), B, 0, stream>>>(deg, 0.f, (size_t)N);
    fill_f32<<<g1((size_t)2 * N, B), B, 0, stream>>>(easum, 0.f, (size_t)2 * N);
    deg_easum_kernel<<<g1(E, B), B, 0, stream>>>(dst0, ea, deg, easum, E);
    mean_ea_kernel<<<g1((size_t)2 * N, B), B, 0, stream>>>(deg, easum, meanea, N);

    // ---- bf16 conversions + weight transposes + edge-attn coefficients
    cvt_bf16_kernel<<<g1((size_t)N * 128, B), B, 0, stream>>>(x, xb, (size_t)N * 128);
    prep_kernel<<<1, 256, 0, stream>>>(W1, We1, att_e1, W2, We2, att_e2, W1t, W2t, P);

    // ---- layer 1: xh1 = x @ W1   (20000x128x256, bf16 WMMA, f32 accum)
    {
        int tiles = (N / 16) * (256 / 16);
        gemm_bf16_wmma<<<(tiles + 7) / 8, 256, 0, stream>>>(xb, W1t, xh1, N, 256, 128);
    }
    att_node_kernel<<<g1((size_t)N * 2, B), B, 0, stream>>>(xh1, att_s1, att_d1, asrc1, adst1, N, 2, 128);

    fill_f32<<<g1((size_t)2 * N, B), B, 0, stream>>>(amax1, -INFINITY, (size_t)2 * N);
    edge_alpha_max_kernel<<<g1((size_t)E2 * 2, B), B, 0, stream>>>(src0, dst0, ea, meanea,
                                                                   asrc1, adst1, P, alpha1, amax1, E, E2, 2);
    fill_f32<<<g1((size_t)2 * N, B), B, 0, stream>>>(denom1, 0.f, (size_t)2 * N);
    edge_exp_sum_kernel<<<g1((size_t)E2 * 2, B), B, 0, stream>>>(dst0, alpha1, amax1, denom1, E, E2, 2);
    fill_f32<<<g1((size_t)N * 256, B), B, 0, stream>>>(out1, 0.f, (size_t)N * 256);
    edge_aggregate_kernel<<<E2, 256, 0, stream>>>(src0, dst0, xh1, alpha1, denom1, out1, E, 2, 128);
    bias_relu_bf16_kernel<<<g1((size_t)N * 256, B), B, 0, stream>>>(out1, b1, hb, (size_t)N * 256, 256);

    // ---- layer 2: xh2 = h @ W2   (20000x256x128, bf16 WMMA)
    {
        int tiles = (N / 16) * (128 / 16);
        gemm_bf16_wmma<<<(tiles + 7) / 8, 256, 0, stream>>>(hb, W2t, xh2, N, 128, 256);
    }
    att_node_kernel<<<g1((size_t)N, B), B, 0, stream>>>(xh2, att_s2, att_d2, asrc2, adst2, N, 1, 128);

    fill_f32<<<g1((size_t)N, B), B, 0, stream>>>(amax2, -INFINITY, (size_t)N);
    edge_alpha_max_kernel<<<g1((size_t)E2, B), B, 0, stream>>>(src0, dst0, ea, meanea,
                                                               asrc2, adst2, P + 4, alpha2, amax2, E, E2, 1);
    fill_f32<<<g1((size_t)N, B), B, 0, stream>>>(denom2, 0.f, (size_t)N);
    edge_exp_sum_kernel<<<g1((size_t)E2, B), B, 0, stream>>>(dst0, alpha2, amax2, denom2, E, E2, 1);
    fill_f32<<<g1((size_t)N * 128, B), B, 0, stream>>>(out2, 0.f, (size_t)N * 128);
    edge_aggregate_kernel<<<E2, 128, 0, stream>>>(src0, dst0, xh2, alpha2, denom2, out2, E, 1, 128);
    bias_add_kernel<<<g1((size_t)N * 128, B), B, 0, stream>>>(out2, b2, (size_t)N * 128, 128);

    // ---- SimpleConv(mean over ORIGINAL edges) + residual + relu
    fill_f32<<<g1((size_t)N * 128, B), B, 0, stream>>>(acc, 0.f, (size_t)N * 128);
    scatter_sum_kernel<<<E, 128, 0, stream>>>(src0, dst0, out2, acc, 128);
    final_kernel<<<g1((size_t)N * 128, B), B, 0, stream>>>(acc, deg, x, out, (size_t)N * 128, 128);
}
